// MultiHeadSelfAttention_88914412962049
// MI455X (gfx1250) — matrix operand order repaired and checked by reading, not compiled, not run on hardware
//
#include <hip/hip_runtime.h>
#include <hip/hip_bf16.h>

// ---------------------------------------------------------------------------
// Types
// ---------------------------------------------------------------------------
typedef __attribute__((ext_vector_type(16))) __bf16 v16bf;
typedef __attribute__((ext_vector_type(8)))  __bf16 v8bf;
typedef __attribute__((ext_vector_type(4)))  __bf16 v4bf;
typedef __attribute__((ext_vector_type(8)))  float  v8f;

#define NEG_BIG (-3.0e38f)

// Problem constants (fixed by the reference)
#define B_SZ   4
#define SEQ    2048
#define DMODEL 1024
#define HEADS  16
#define DK     64

__device__ __forceinline__ v16bf frag_cat(v8bf lo, v8bf hi) {
  return __builtin_shufflevector(lo, hi, 0,1,2,3,4,5,6,7,8,9,10,11,12,13,14,15);
}

// A-operand fragment of a 16x32 bf16 tile. Source memory is [row][k] with
// `stride` elements per row. ISA 7.12.2 layout: lane&15 = M row,
// element j -> k = (j>>3)*16 + (lane>>4)*8 + (j&7)  (two contiguous 8-elem runs)
__device__ __forceinline__ v16bf load_frag_a(const __bf16* base, int row0,
                                             int stride, int koff) {
  const int lane = threadIdx.x & 31;
  const int half = lane >> 4;
  const int r    = lane & 15;
  const __bf16* p = base + (size_t)(row0 + r) * stride + koff + half * 8;
  v8bf lo = *(const v8bf*)(p);        // k = koff + half*8 + 0..7
  v8bf hi = *(const v8bf*)(p + 16);   // k = koff + 16 + half*8 + 0..7
  return frag_cat(lo, hi);
}

// B-operand fragment of a 32x16 bf16 tile. Source memory is [n][k] (one row
// per output column n) with `stride` elems/row.
// ISA layout: lane&15 = N column, element j -> k = (j>>3)*16 + (lane>>4)*8 + (j&7)
__device__ __forceinline__ v16bf load_frag_b(const __bf16* base, int col0,
                                             int stride, int koff) {
  const int lane = threadIdx.x & 31;
  const int half = lane >> 4;
  const int n    = lane & 15;
  const __bf16* p = base + (size_t)(col0 + n) * stride + koff + half * 8;
  v8bf lo = *(const v8bf*)(p);        // k = koff + half*8 + 0..7
  v8bf hi = *(const v8bf*)(p + 16);   // k = koff + 16 + half*8 + 0..7
  return frag_cat(lo, hi);
}

__device__ __forceinline__ v8f wmma_bf16(v16bf a, v16bf b, v8f c) {
  return __builtin_amdgcn_wmma_f32_16x16x32_bf16(
      /*neg_a=*/false, a, /*neg_b=*/false, b,
      /*c_mod=*/(short)0, c, /*reuse_a=*/false, /*reuse_b=*/false);
}

// --- CDNA5 async global->LDS (ASYNCcnt path, no VGPR round trip) -----------
// LDS address is taken from the vdst VGPR; low 32 bits of a generic __shared__ pointer
// are the LDS-aperture offset per the flat-address mapping.
__device__ __forceinline__ void async_ld_b128(__bf16* lds_p, const __bf16* g_p) {
  unsigned           lo = (unsigned)(unsigned long long)lds_p;
  unsigned long long ga = (unsigned long long)g_p;
  asm volatile("global_load_async_to_lds_b128 %0, %1, off"
               :: "v"(lo), "v"(ga) : "memory");
}
__device__ __forceinline__ void wait_async0() {
  asm volatile("s_wait_asynccnt 0x0" ::: "memory");
}

// ---------------------------------------------------------------------------
// f32 -> bf16 conversion (vectorized x4)
// ---------------------------------------------------------------------------
__global__ __launch_bounds__(256) void cvt_f32_bf16(const float* __restrict__ src,
                                                    __bf16* __restrict__ dst,
                                                    int n4) {
  int i = blockIdx.x * 256 + threadIdx.x;
  if (i < n4) {
    float4 f = ((const float4*)src)[i];
    v4bf o;
    o[0] = (__bf16)f.x; o[1] = (__bf16)f.y;
    o[2] = (__bf16)f.z; o[3] = (__bf16)f.w;
    ((v4bf*)dst)[i] = o;
  }
}

// ---------------------------------------------------------------------------
// bf16 GEMM:  C[M,N] = A[M,K] * B[N,K]^T   (torch Linear: y = x @ W^T)
// Block tile 128x128, 8 wave32 waves, each wave 64x32 (4x2 WMMA accums).
// Double-buffered LDS fed by async global->LDS DMA; one barrier per K-step.
// mode 0: write f32 row-major [M,N] (final output projection)
// mode 1: write bf16 split-head [B,H,S,DK]  (Q/K/V projections)
// ---------------------------------------------------------------------------
__global__ __launch_bounds__(256) void gemm_bf16(const __bf16* __restrict__ A,
                                                 const __bf16* __restrict__ Bw,
                                                 void* __restrict__ Out,
                                                 int M, int N, int K, int mode) {
  __shared__ __bf16 lds_a[2][128 * 40];  // padded stride 40 -> conflict-free
  __shared__ __bf16 lds_b[2][128 * 40];

  const int m0   = blockIdx.x * 128;
  const int n0   = blockIdx.y * 128;
  const int tid  = threadIdx.x;
  const int wave = tid >> 5;
  const int wm   = (wave & 1) * 64;    // wave M offset within block
  const int wn   = (wave >> 1) * 32;   // wave N offset within block

  // per-thread staging coordinates: 512 8-elem chunks per 128x32 tile
  const int r0 = tid >> 2,          c0 = (tid & 3) * 8;           // chunk 0
  const int r1 = (tid + 256) >> 2,  c1 = ((tid + 256) & 3) * 8;   // chunk 1

  v8f acc[4][2] = {};

  // prologue: stage tile 0 into buffer 0
  async_ld_b128(&lds_a[0][r0 * 40 + c0], &A [(size_t)(m0 + r0) * K + c0]);
  async_ld_b128(&lds_b[0][r0 * 40 + c0], &Bw[(size_t)(n0 + r0) * K + c0]);
  async_ld_b128(&lds_a[0][r1 * 40 + c1], &A [(size_t)(m0 + r1) * K + c1]);
  async_ld_b128(&lds_b[0][r1 * 40 + c1], &Bw[(size_t)(n0 + r1) * K + c1]);

  int cur = 0;
  for (int k0 = 0; k0 < K; k0 += 32) {
    wait_async0();        // own async ops done (covers tile k0)
    __syncthreads();      // everyone's tile k0 visible

    if (k0 + 32 < K) {    // stage next tile into the other buffer
      int nb = cur ^ 1;
      async_ld_b128(&lds_a[nb][r0 * 40 + c0], &A [(size_t)(m0 + r0) * K + k0 + 32 + c0]);
      async_ld_b128(&lds_b[nb][r0 * 40 + c0], &Bw[(size_t)(n0 + r0) * K + k0 + 32 + c0]);
      async_ld_b128(&lds_a[nb][r1 * 40 + c1], &A [(size_t)(m0 + r1) * K + k0 + 32 + c1]);
      async_ld_b128(&lds_b[nb][r1 * 40 + c1], &Bw[(size_t)(n0 + r1) * K + k0 + 32 + c1]);
    }

    v16bf af[4], bfr[2];
    #pragma unroll
    for (int mt = 0; mt < 4; ++mt) af[mt]  = load_frag_a(lds_a[cur], wm + mt * 16, 40, 0);
    #pragma unroll
    for (int nt = 0; nt < 2; ++nt) bfr[nt] = load_frag_b(lds_b[cur], wn + nt * 16, 40, 0);
    #pragma unroll
    for (int mt = 0; mt < 4; ++mt)
      #pragma unroll
      for (int nt = 0; nt < 2; ++nt)
        acc[mt][nt] = wmma_bf16(af[mt], bfr[nt], acc[mt][nt]);

    cur ^= 1;
  }

  // Epilogue. C/D layout: row = v + (lane>>4)*8, col = lane&15.
  const int lane = tid & 31, half = lane >> 4, colL = lane & 15;
  #pragma unroll
  for (int mt = 0; mt < 4; ++mt)
    #pragma unroll
    for (int nt = 0; nt < 2; ++nt)
      #pragma unroll
      for (int v = 0; v < 8; ++v) {
        int grow = m0 + wm + mt * 16 + v + half * 8;
        int gcol = n0 + wn + nt * 16 + colL;
        float val = acc[mt][nt][v];
        if (mode == 0) {
          ((float*)Out)[(size_t)grow * N + gcol] = val;
        } else {
          int b = grow >> 11, s = grow & (SEQ - 1);   // grow = b*SEQ + s
          int h = gcol >> 6,  d = gcol & (DK - 1);    // gcol = h*DK + d
          ((__bf16*)Out)[(((size_t)(b * HEADS + h) * SEQ) + s) * DK + d] =
              (__bf16)val;
        }
      }
}

// ---------------------------------------------------------------------------
// Flash attention over bf16 Q/K/V [B,H,S,DK] with causal mask.
// Block = 128 query rows of one (b,h); 8 waves x 16 rows; 64-key tiles.
// K tile staged via async DMA; V tile register-transposed into LDS.
// ---------------------------------------------------------------------------
__global__ __launch_bounds__(256) void attn_kernel(const __bf16* __restrict__ Qb,
                                                   const __bf16* __restrict__ Kb,
                                                   const __bf16* __restrict__ Vb,
                                                   __bf16* __restrict__ Ctx) {
  __shared__ __bf16 kt[64 * 72];        // K tile   [key][dk]
  __shared__ __bf16 vt[64 * 72];        // V tile^T [dk][key]
  __shared__ __bf16 pb[8 * 16 * 72];    // per-wave P tiles [16 q][64 key]

  const int q0    = blockIdx.x * 128;
  const int bh    = blockIdx.y;                 // b*HEADS + h
  const int tid   = threadIdx.x;
  const int wave  = tid >> 5;
  const int lane  = tid & 31, half = lane >> 4, colL = lane & 15;
  const int qbase = q0 + wave * 16;
  const size_t base = (size_t)bh * SEQ * DK;
  const float scale = 0.125f;                   // 1/sqrt(64)

  // Q A-fragments straight from global memory (layout-compatible): 16 rows x 64
  v16bf qa[2];
  #pragma unroll
  for (int kc = 0; kc < 2; ++kc)
    qa[kc] = load_frag_a(Qb + base, qbase, DK, kc * 32);

  float mrow[8], lrow[8];
  #pragma unroll
  for (int v = 0; v < 8; ++v) { mrow[v] = NEG_BIG; lrow[v] = 0.f; }
  v8f oacc[4] = {};

  __bf16* pw = &pb[wave * 16 * 72];

  for (int jb = 0; jb < q0 + 128; jb += 64) {
    // --- stage K tile [key][dk] via async DMA: 512 chunks, 2 per thread ---
    #pragma unroll
    for (int i = 0; i < 2; ++i) {
      int c = tid + i * 256;
      int row = c >> 3, col = (c & 7) * 8;
      async_ld_b128(&kt[row * 72 + col],
                    &Kb[base + (size_t)(jb + row) * DK + col]);
    }
    // --- stage V tile transposed [dk][key]: 1024 4-elem chunks, 4/thread ---
    #pragma unroll
    for (int i = 0; i < 4; ++i) {
      int c = tid + i * 256;
      int row = c >> 4, col4 = (c & 15) * 4;
      const __bf16* vp = &Vb[base + (size_t)(jb + row) * DK + col4];
      __bf16 t0 = vp[0], t1 = vp[1], t2 = vp[2], t3 = vp[3];
      vt[(col4 + 0) * 72 + row] = t0;
      vt[(col4 + 1) * 72 + row] = t1;
      vt[(col4 + 2) * 72 + row] = t2;
      vt[(col4 + 3) * 72 + row] = t3;
    }
    if (jb + 64 < q0 + 128) {   // prefetch next key tile (global_prefetch_b8)
      __builtin_prefetch(&Kb[base + (size_t)(jb + 64 + (tid >> 2)) * DK], 0, 1);
      __builtin_prefetch(&Vb[base + (size_t)(jb + 64 + (tid >> 2)) * DK], 0, 1);
    }
    wait_async0();
    __syncthreads();

    // --- scores S = Q K^T for 4 key subtiles (dk = 2 x K32) ---
    v8f sf[4];
    #pragma unroll
    for (int nt = 0; nt < 4; ++nt) {
      v16bf bk0 = load_frag_b(kt, nt * 16, 72, 0);
      v16bf bk1 = load_frag_b(kt, nt * 16, 72, 32);
      v8f s = {};
      s = wmma_bf16(qa[0], bk0, s);
      s = wmma_bf16(qa[1], bk1, s);
      sf[nt] = s;
    }

    // --- scale + causal mask + row max ---
    float mloc[8];
    #pragma unroll
    for (int v = 0; v < 8; ++v) mloc[v] = NEG_BIG;
    #pragma unroll
    for (int nt = 0; nt < 4; ++nt) {
      int key = jb + nt * 16 + colL;
      #pragma unroll
      for (int v = 0; v < 8; ++v) {
        int q = qbase + v + half * 8;
        float val = sf[nt][v] * scale;
        if (key > q) val = NEG_BIG;
        sf[nt][v] = val;
        mloc[v] = fmaxf(mloc[v], val);
      }
    }
    // reduce max across the 16 columns (xor<16 stays within the half-wave)
    #pragma unroll
    for (int off = 1; off < 16; off <<= 1)
      #pragma unroll
      for (int v = 0; v < 8; ++v)
        mloc[v] = fmaxf(mloc[v], __shfl_xor(mloc[v], off, 32));

    // --- online softmax update ---
    float alpha[8];
    #pragma unroll
    for (int v = 0; v < 8; ++v) {
      float mn = fmaxf(mrow[v], mloc[v]);
      alpha[v] = __expf(mrow[v] - mn);
      mrow[v]  = mn;
      lrow[v] *= alpha[v];
    }
    #pragma unroll
    for (int dt = 0; dt < 4; ++dt)
      #pragma unroll
      for (int v = 0; v < 8; ++v) oacc[dt][v] *= alpha[v];

    // p = exp(s - m): accumulate per-lane partial row sum, spill bf16 P to LDS
    #pragma unroll
    for (int nt = 0; nt < 4; ++nt)
      #pragma unroll
      for (int v = 0; v < 8; ++v) {
        float p = __expf(sf[nt][v] - mrow[v]);
        lrow[v] += p;
        pw[(v + half * 8) * 72 + nt * 16 + colL] = (__bf16)p;
      }

    // same-wave LDS RAW: drain DScnt (CDNA5 split counter) before re-reading P
    asm volatile("s_wait_dscnt 0x0" ::: "memory");

    // --- O += P V : P as A-operand, V^T rows as B-operand ---
    #pragma unroll
    for (int kc = 0; kc < 2; ++kc) {
      v16bf ap = load_frag_a(pw, 0, 72, kc * 32);
      #pragma unroll
      for (int dt = 0; dt < 4; ++dt) {
        v16bf bv = load_frag_b(vt, dt * 16, 72, kc * 32);
        oacc[dt] = wmma_bf16(ap, bv, oacc[dt]);
      }
    }
    __syncthreads();   // protect kt/vt before next tile's staging
  }

  // --- finalize: reduce l across columns, normalize, merge heads ---
  #pragma unroll
  for (int off = 1; off < 16; off <<= 1)
    #pragma unroll
    for (int v = 0; v < 8; ++v) lrow[v] += __shfl_xor(lrow[v], off, 32);

  const int b = bh >> 4, h = bh & (HEADS - 1);
  #pragma unroll
  for (int dt = 0; dt < 4; ++dt)
    #pragma unroll
    for (int v = 0; v < 8; ++v) {
      int srow = qbase + v + half * 8;
      int dcol = dt * 16 + colL;
      float val = oacc[dt][v] / lrow[v];
      Ctx[((size_t)(b * SEQ + srow)) * DMODEL + h * DK + dcol] = (__bf16)val;
    }
}

// ---------------------------------------------------------------------------
// Launch
// ---------------------------------------------------------------------------
extern "C" void kernel_launch(void* const* d_in, const int* in_sizes, int n_in,
                              void* d_out, int out_size, void* d_ws, size_t ws_size,
                              hipStream_t stream) {
  const float* x  = (const float*)d_in[0];
  const float* Wq = (const float*)d_in[1];
  const float* Wk = (const float*)d_in[2];
  const float* Wv = (const float*)d_in[3];
  const float* Wo = (const float*)d_in[4];

  char* ws = (char*)d_ws;
  const size_t MB = 1024ull * 1024ull;
  __bf16* xb  = (__bf16*)(ws + 0 * MB);    // 16 MB  (8192x1024 bf16)
  __bf16* wqb = (__bf16*)(ws + 16 * MB);   // 2 MB each
  __bf16* wkb = (__bf16*)(ws + 18 * MB);
  __bf16* wvb = (__bf16*)(ws + 20 * MB);
  __bf16* wob = (__bf16*)(ws + 22 * MB);
  __bf16* Qb  = (__bf16*)(ws + 24 * MB);   // 16 MB [B,H,S,DK]
  __bf16* Kb  = (__bf16*)(ws + 40 * MB);
  __bf16* Vb  = (__bf16*)(ws + 56 * MB);
  __bf16* Ctx = (__bf16*)(ws + 72 * MB);   // 16 MB [B,S,DMODEL]

  const int ND = B_SZ * SEQ * DMODEL;      // 8,388,608
  const int NW = DMODEL * DMODEL;          // 1,048,576

  cvt_f32_bf16<<<(ND / 4 + 255) / 256, 256, 0, stream>>>(x,  xb,  ND / 4);
  cvt_f32_bf16<<<(NW / 4 + 255) / 256, 256, 0, stream>>>(Wq, wqb, NW / 4);
  cvt_f32_bf16<<<(NW / 4 + 255) / 256, 256, 0, stream>>>(Wk, wkb, NW / 4);
  cvt_f32_bf16<<<(NW / 4 + 255) / 256, 256, 0, stream>>>(Wv, wvb, NW / 4);
  cvt_f32_bf16<<<(NW / 4 + 255) / 256, 256, 0, stream>>>(Wo, wob, NW / 4);

  const int M = B_SZ * SEQ;                // 8192
  dim3 gg(M / 128, DMODEL / 128);          // 64 x 8
  gemm_bf16<<<gg, 256, 0, stream>>>(xb, wqb, Qb, M, DMODEL, DMODEL, 1);
  gemm_bf16<<<gg, 256, 0, stream>>>(xb, wkb, Kb, M, DMODEL, DMODEL, 1);
  gemm_bf16<<<gg, 256, 0, stream>>>(xb, wvb, Vb, M, DMODEL, DMODEL, 1);

  attn_kernel<<<dim3(SEQ / 128, B_SZ * HEADS), 256, 0, stream>>>(Qb, Kb, Vb, Ctx);

  gemm_bf16<<<gg, 256, 0, stream>>>(Ctx, wob, d_out, M, DMODEL, DMODEL, 0);
}
